// RecursiveMamba130m_v28_57105885167720
// MI455X (gfx1250) — compile-verified, matrix operand order checked by reading, not run on hardware
//
#include <hip/hip_runtime.h>
#include <hip/hip_bf16.h>

typedef __attribute__((ext_vector_type(16))) __bf16 v16bf;
typedef __attribute__((ext_vector_type(8)))  float  v8f;
typedef __attribute__((ext_vector_type(4)))  float  v4f;

#define L_ 6
#define B_ 16
#define T_ 2048
#define D_ 768
#define V_ 50280
#define M_ 96                    // L_*B_
#define KSTEPS 24                // D_/32
#define NTILES 3143              // ceil(V_/16)
#define WAVES_PER_BLOCK 8

// ---------------------------------------------------------------------------
// Zero a byte region (as u32 words; region sizes are multiples of 4).
__global__ void k_zero_u32(unsigned int* p, int nwords) {
    int i = blockIdx.x * blockDim.x + threadIdx.x;
    if (i < nwords) p[i] = 0u;
}

// ---------------------------------------------------------------------------
// present[b, v] = 1 for every v in input_ids[b, :]. Races write the same value.
__global__ void k_scatter_present(const int* __restrict__ input_ids,
                                  unsigned char* __restrict__ present) {
    int i = blockIdx.x * blockDim.x + threadIdx.x;
    if (i < B_ * T_) {
        int b = i / T_;
        int v = input_ids[i];
        if (v >= 0 && v < V_) present[(size_t)b * V_ + v] = 1;
    }
}

// ---------------------------------------------------------------------------
// Gather h[l,b,:] = x_normed[l, b, ans_starts[b]-1, :], split each fp32 into
// bf16 hi + bf16 lo residual, and store in the WMMA 16-bit A-operand layout:
//   tile t = l (rows m = l*16 + b, r = b), per K-chunk kk (32 values):
//     lane (r or r+16) holds 16 contiguous bf16 covering K runs
//     {0..7,16..23} (lanes 0-15) / {8..15,24..31} (lanes 16-31).
// Element offset: ((t*KSTEPS + kk)*32 + lane)*16 + idx.
__global__ void k_prep_h(const float* __restrict__ x_normed,
                         const int* __restrict__ ans_starts,
                         __bf16* __restrict__ h_hi,
                         __bf16* __restrict__ h_lo) {
    int m = blockIdx.x;          // 0..95
    int l = m >> 4;
    int b = m & 15;
    int as = ans_starts[b];
    int tpos = as - 1;
    if (tpos < 0) tpos = 0;
    if (tpos >= T_) tpos = T_ - 1;
    const float* src = x_normed + (((size_t)l * B_ + b) * T_ + tpos) * D_;
    for (int k = threadIdx.x; k < D_; k += blockDim.x) {
        float xv = src[k];
        __bf16 hi = (__bf16)xv;
        __bf16 lo = (__bf16)(xv - (float)hi);
        int kk   = k >> 5;
        int ko   = k & 31;
        int half = ko >> 4;       // which 16-run inside the 32-chunk
        int sub  = ko & 15;
        int lane = (sub < 8) ? b : (b + 16);
        int idx  = (sub & 7) + 8 * half;
        size_t off = (((size_t)l * KSTEPS + kk) * 32 + lane) * 16 + idx;
        h_hi[off] = hi;
        h_lo[off] = lo;
    }
}

// ---------------------------------------------------------------------------
// Main GEMM: logits[m, v] = sum_d h[m,d] * W[v,d], m in [0,96), v in [0,V).
// One wave per 16-column N-tile; 6 M-tiles (all 96 rows) accumulated in
// registers so every W element is read from HBM exactly once.
// Split-precision: acc += Ahi*Bhi + Ahi*Blo + Alo*Bhi (f32 accumulate).
__global__ void __launch_bounds__(32 * WAVES_PER_BLOCK)
k_gemm_logits(const float* __restrict__ W,       // (V, D) row-major fp32
              const __bf16* __restrict__ h_hi,   // swizzled A operands
              const __bf16* __restrict__ h_lo,
              float* __restrict__ logits) {      // (96, V)
    const int lane = threadIdx.x & 31;
    const int wave = threadIdx.x >> 5;
    const int tile = blockIdx.x * WAVES_PER_BLOCK + wave;
    if (tile >= NTILES) return;   // wave-uniform: EXEC stays all-ones for WMMA

    const int v0  = tile * 16;
    const int col = lane & 15;
    int vrow = v0 + col;
    vrow = (vrow < V_) ? vrow : (V_ - 1);        // branchless clamp for loads
    // Lanes 0-15 supply K runs {0..7,16..23}; lanes 16-31 supply {8..15,24..31}.
    const float* wrow = W + (size_t)vrow * D_ + ((lane < 16) ? 0 : 8);

    v8f acc[6];
#pragma unroll
    for (int t = 0; t < 6; ++t)
#pragma unroll
        for (int r = 0; r < 8; ++r) acc[t][r] = 0.0f;

    for (int kk = 0; kk < KSTEPS; ++kk) {
        const float* p = wrow + kk * 32;
        // Prefetch next K-chunk of this W row (streaming hint).
        if (kk + 1 < KSTEPS) __builtin_prefetch(p + 32, 0, 0);

        v4f f0 = *(const v4f*)(p);        // K run base + 0..3
        v4f f1 = *(const v4f*)(p + 4);    // K run base + 4..7
        v4f f2 = *(const v4f*)(p + 16);   // K run base+16 .. +19
        v4f f3 = *(const v4f*)(p + 20);   // K run base+20 .. +23

        v16bf bhi, blo;
#pragma unroll
        for (int j = 0; j < 4; ++j) {
            float x;
            __bf16 h;
            x = f0[j]; h = (__bf16)x; bhi[j]      = h; blo[j]      = (__bf16)(x - (float)h);
            x = f1[j]; h = (__bf16)x; bhi[4 + j]  = h; blo[4 + j]  = (__bf16)(x - (float)h);
            x = f2[j]; h = (__bf16)x; bhi[8 + j]  = h; blo[8 + j]  = (__bf16)(x - (float)h);
            x = f3[j]; h = (__bf16)x; bhi[12 + j] = h; blo[12 + j] = (__bf16)(x - (float)h);
        }

#pragma unroll
        for (int t = 0; t < 6; ++t) {
            size_t off = (((size_t)t * KSTEPS + kk) * 32 + lane) * 16;
            v16bf ahi = *(const v16bf*)(h_hi + off);
            v16bf alo = *(const v16bf*)(h_lo + off);
            acc[t] = __builtin_amdgcn_wmma_f32_16x16x32_bf16(
                false, ahi, false, bhi, (short)0, acc[t], false, false);
            acc[t] = __builtin_amdgcn_wmma_f32_16x16x32_bf16(
                false, ahi, false, blo, (short)0, acc[t], false, false);
            acc[t] = __builtin_amdgcn_wmma_f32_16x16x32_bf16(
                false, alo, false, bhi, (short)0, acc[t], false, false);
        }
    }

    // C/D layout: VGPR r -> lanes 0-15: (M=r, N=lane); lanes 16-31: (M=r+8, N=lane-16)
    const int n = v0 + col;
    const bool ok = (n < V_);
    const int mofs = (lane < 16) ? 0 : 8;
#pragma unroll
    for (int t = 0; t < 6; ++t) {
#pragma unroll
        for (int r = 0; r < 8; ++r) {
            if (ok) logits[(size_t)(t * 16 + mofs + r) * V_ + n] = acc[t][r];
        }
    }
}

// ---------------------------------------------------------------------------
// Per-row masked reduction: max, first-argmax, sum(exp(x - max)) over present.
__global__ void k_row_reduce(const float* __restrict__ logits,
                             const unsigned char* __restrict__ present,
                             float* __restrict__ rmax,
                             float* __restrict__ rsum,
                             int* __restrict__ rarg) {
    __shared__ float smax[256];
    __shared__ int   sidx[256];
    const int m = blockIdx.x;
    const int b = m & 15;
    const float* row = logits + (size_t)m * V_;
    const unsigned char* pres = present + (size_t)b * V_;

    float mx = -INFINITY;
    int   mi = V_;   // sentinel > any valid index (argmax tie -> lowest index)
    for (int v = threadIdx.x; v < V_; v += blockDim.x) {
        if (pres[v]) {
            float x = row[v];
            if (x > mx || (x == mx && v < mi)) { mx = x; mi = v; }
        }
    }
    smax[threadIdx.x] = mx;
    sidx[threadIdx.x] = mi;
    __syncthreads();
    for (int s = 128; s > 0; s >>= 1) {
        if (threadIdx.x < s) {
            float o = smax[threadIdx.x + s];
            int  oi = sidx[threadIdx.x + s];
            if (o > smax[threadIdx.x] ||
                (o == smax[threadIdx.x] && oi < sidx[threadIdx.x])) {
                smax[threadIdx.x] = o;
                sidx[threadIdx.x] = oi;
            }
        }
        __syncthreads();
    }
    const float gmax = smax[0];
    const int   garg = sidx[0];
    __syncthreads();

    float s = 0.0f;
    for (int v = threadIdx.x; v < V_; v += blockDim.x) {
        if (pres[v]) s += expf(row[v] - gmax);
    }
    smax[threadIdx.x] = s;
    __syncthreads();
    for (int st = 128; st > 0; st >>= 1) {
        if (threadIdx.x < st) smax[threadIdx.x] += smax[threadIdx.x + st];
        __syncthreads();
    }
    if (threadIdx.x == 0) {
        rmax[m] = gmax;
        rsum[m] = smax[0];
        rarg[m] = garg;
    }
}

// ---------------------------------------------------------------------------
// Final scalars: avg_loss, avg_acc, final_acc.
__global__ void k_finalize(const float* __restrict__ logits,
                           const unsigned char* __restrict__ present,
                           const int* __restrict__ chain_targets, // (B, L)
                           const int* __restrict__ ans_starts,
                           const float* __restrict__ rmax,
                           const float* __restrict__ rsum,
                           const int* __restrict__ rarg,
                           float* __restrict__ out) {
    __shared__ float nllv[M_], accv[M_], valv[M_];
    const int tid = threadIdx.x;
    if (tid < M_) {
        int l = tid >> 4;
        int b = tid & 15;
        int tgt = chain_targets[b * L_ + l];
        int tg  = tgt < 0 ? 0 : (tgt >= V_ ? V_ - 1 : tgt);
        int as  = ans_starts[b];
        bool inr   = (as >= 1) && (as < T_);
        bool valid = inr && (tgt >= 0) && (tgt < V_) &&
                     (present[(size_t)b * V_ + tg] != 0);
        float logp = logits[(size_t)tid * V_ + tg] - rmax[tid] - logf(rsum[tid]);
        float nll  = -logp;
        float acc  = (rarg[tid] == tgt) ? 1.0f : 0.0f;
        nllv[tid] = valid ? nll : 0.0f;
        accv[tid] = valid ? acc : 0.0f;
        valv[tid] = valid ? 1.0f : 0.0f;
    }
    __syncthreads();
    if (tid == 0) {
        float avl = 0.0f, ava = 0.0f, fin = 0.0f, nvl = 0.0f;
        for (int l = 0; l < L_; ++l) {
            float cnt = 0.0f, ls = 0.0f, ac = 0.0f;
            for (int b = 0; b < B_; ++b) {
                int m = l * 16 + b;
                cnt += valv[m];
                ls  += nllv[m];
                ac  += accv[m];
            }
            float ll = ls / fmaxf(cnt, 1.0f);
            float la = ac / fmaxf(cnt, 1.0f);
            if (cnt > 0.0f) { avl += ll; ava += la; nvl += 1.0f; }
            if (l == L_ - 1) fin = la;
        }
        float d = fmaxf(nvl, 1.0f);
        out[0] = avl / d;
        out[1] = ava / d;
        out[2] = fin;
    }
}

// ---------------------------------------------------------------------------
extern "C" void kernel_launch(void* const* d_in, const int* in_sizes, int n_in,
                              void* d_out, int out_size, void* d_ws, size_t ws_size,
                              hipStream_t stream) {
    const float* x_normed      = (const float*)d_in[0]; // (L,B,T,D)
    const float* lm_head_w     = (const float*)d_in[1]; // (V,D)
    const int*   input_ids     = (const int*)d_in[2];   // (B,T)
    const int*   chain_targets = (const int*)d_in[3];   // (B,L)
    const int*   ans_starts    = (const int*)d_in[4];   // (B,)
    float* out = (float*)d_out;

    // Workspace layout (all offsets 256B-aligned).
    char* ws = (char*)d_ws;
    const size_t hA_elems = (size_t)M_ * D_;               // 73728 bf16
    __bf16* h_hi = (__bf16*)(ws);                          // 147456 B
    __bf16* h_lo = (__bf16*)(ws + 147456);                 // 147456 B
    unsigned char* present = (unsigned char*)(ws + 294912);// 804480 B
    float* rmax = (float*)(ws + 1099520);                  // 384 B (pad 512)
    float* rsum = (float*)(ws + 1100032);
    int*   rarg = (int*)  (ws + 1100544);
    float* logits = (float*)(ws + 1101824);                // 96*V_*4 = 19307520 B
    (void)hA_elems; (void)ws_size; (void)n_in; (void)in_sizes; (void)out_size;

    // 1) zero present (804480 bytes = 201120 words)
    {
        int nwords = (B_ * V_) / 4;
        int grid = (nwords + 255) / 256;
        k_zero_u32<<<grid, 256, 0, stream>>>((unsigned int*)present, nwords);
    }
    // 2) scatter present from input_ids
    {
        int n = B_ * T_;
        k_scatter_present<<<(n + 255) / 256, 256, 0, stream>>>(input_ids, present);
    }
    // 3) gather + bf16 hi/lo split + WMMA A-layout swizzle of h
    k_prep_h<<<M_, 256, 0, stream>>>(x_normed, ans_starts, h_hi, h_lo);

    // 4) main WMMA GEMM: one wave per 16-column tile of V
    {
        int grid = (NTILES + WAVES_PER_BLOCK - 1) / WAVES_PER_BLOCK; // 393
        k_gemm_logits<<<grid, 32 * WAVES_PER_BLOCK, 0, stream>>>(
            lm_head_w, h_hi, h_lo, logits);
    }

    // 5) per-row masked max / argmax / sumexp
    k_row_reduce<<<M_, 256, 0, stream>>>(logits, present, rmax, rsum, rarg);

    // 6) finalize the three scalars
    k_finalize<<<1, 128, 0, stream>>>(logits, present, chain_targets, ans_starts,
                                      rmax, rsum, rarg, out);
}
